// MixingAttention_13735305412885
// MI455X (gfx1250) — matrix operand order repaired and checked by reading, not compiled, not run on hardware
//
#include <hip/hip_runtime.h>
#include <hip/hip_bf16.h>

// ---------------- problem constants ----------------
#define B_      64
#define NTOK    784
#define DIM     256
#define N2_     4
#define HS_     28
#define WS_     7
#define M_      8
#define D_      2
#define HW_     196
#define C_      32
#define KC      392       // HW_*D_   (K of the big GEMM)
#define KCP     416       // K padded to 13*32
#define NG      38416     // HW_*HW_  (N of the big GEMM)
#define NWIN    32        // window rows per block (2 WMMA M-tiles share every B load)

// ---------------- LDS layout (dynamic, 231 KB) ----------------
#define SMEM_SS    (HW_ * 512 * 2)     // bf16 score tile [i=196][j=16][w=32]   = 200704 B
#define SMEM_SA    (NWIN * KCP * 2)    // bf16 A matrix   [w=32][K=416]         = 26624 B
#define SMEM_ST    (1024 * 4)          // softmax stats: max[512] + rcp[512]    = 4096 B
#define SMEM_TOTAL (SMEM_SS + SMEM_SA + SMEM_ST)

#define WS_NEED    ((size_t)NG * KCP * 2)   // 31,962,112 B: transposed bf16 Wg

typedef __attribute__((ext_vector_type(16))) __bf16 v16bf;
typedef __attribute__((ext_vector_type(8)))  __bf16 v8bf;
typedef __attribute__((ext_vector_type(8)))  float  v8f;

__device__ __forceinline__ __bf16 to_bf16(float f) { return (__bf16)f; }

// A-matrix (16x32 bf16) K offset for element e (0..15) in lane-half `half`:
//   lanes 0-15 hold K=0..15, lanes 16-31 hold K=16..31 (same order as B)
__device__ __forceinline__ int koffA(int e, int half) {
  return (half << 4) + e;
}
// B-matrix (32x16 bf16) K offset: lanes 0-15 hold K=0..15, lanes 16-31 hold K=16..31
__device__ __forceinline__ int koffB(int e, int half) {
  return (half << 4) + e;
}

// ============ Pre-pass: WgT[n][k] = bf16(Wg[k][n]), K zero-padded to 416 ============
__global__ __launch_bounds__(256, 4)
void wg_transpose_kernel(const float* __restrict__ Wg, __bf16* __restrict__ WgT) {
  __shared__ __bf16 tile[32][33];
  const int tx = threadIdx.x;        // 0..31
  const int ty = threadIdx.y;        // 0..7
  const int n0 = blockIdx.x * 32;
  const int k0 = blockIdx.y * 32;
  #pragma unroll
  for (int r = 0; r < 4; ++r) {
    int k = k0 + ty + r * 8;
    int n = n0 + tx;
    float v = (k < KC && n < NG) ? Wg[(size_t)k * NG + n] : 0.0f;
    tile[ty + r * 8][tx] = to_bf16(v);
  }
  __syncthreads();
  #pragma unroll
  for (int r = 0; r < 4; ++r) {
    int nl = ty + r * 8;
    int n  = n0 + nl;
    int k  = k0 + tx;
    if (n < NG)
      WgT[(size_t)n * KCP + k] = tile[tx][nl];
  }
}

// ============ Fused mixing-attention kernel (one block per batch) ============
template <bool TPOSE>
__global__ __launch_bounds__(256, 1)
void mixing_attention_kernel(const float* __restrict__ x,
                             const float* __restrict__ Wc,
                             const float* __restrict__ bc,
                             const float* __restrict__ Wg,
                             const __bf16* __restrict__ WgT,
                             const float* __restrict__ bg,
                             float* __restrict__ out)
{
  extern __shared__ char smem[];
  __bf16* sSb  = (__bf16*)smem;                             // [i][j][w] bf16 scores
  __bf16* sA   = (__bf16*)(smem + SMEM_SS);                 // [w][KCP] bf16
  float*  sMax = (float*)(smem + SMEM_SS + SMEM_SA);        // [512] (col = j*32+w)
  float*  sRcp = sMax + 512;                                // [512]
  float*  sWcT = (float*)smem;                              // [16][256] staged Wc^T (phase A only)

  const int tid  = threadIdx.x;
  const int lane = tid & 31;          // wave32
  const int wave = tid >> 5;          // 8 waves
  const int half = lane >> 4;
  const int l15  = lane & 15;

  const int b = blockIdx.x;           // one batch per block; all 32 windows
  const float* xb = x + (size_t)b * NTOK * DIM;

  // ---------- stage Wc^T into (currently unused) score region ----------
  for (int idx = tid; idx < (M_ * D_) * DIM; idx += 256) {
    int md = idx >> 8, ch = idx & 255;
    sWcT[md * DIM + ch] = Wc[ch * (M_ * D_) + md];
  }
  __syncthreads();

  // ---------- Phase A: compress  w = x @ Wc + bc  -> sA (bf16) ----------
  for (int idx = tid; idx < NWIN * KC; idx += 256) {
    int w = idx / KC, k = idx - w * KC;
    int i = k >> 1, d = k & 1;                  // k = i*2 + d
    int h = i / WS_, wc2 = i - h * WS_;
    int n2 = w >> 3, m = w & 7;
    int tok = h * (N2_ * WS_) + n2 * WS_ + wc2;
    int md  = m * D_ + d;
    const float4* xp4 = (const float4*)(xb + (size_t)tok * DIM);
    const float4* wc4 = (const float4*)(sWcT + md * DIM);
    float acc = bc[md];
    #pragma unroll 4
    for (int q = 0; q < DIM / 4; ++q) {
      float4 xv = xp4[q], wv = wc4[q];
      acc = fmaf(xv.x, wv.x, acc);
      acc = fmaf(xv.y, wv.y, acc);
      acc = fmaf(xv.z, wv.z, acc);
      acc = fmaf(xv.w, wv.w, acc);
    }
    sA[w * KCP + k] = to_bf16(acc);
  }
  for (int idx = tid; idx < NWIN * (KCP - KC); idx += 256) {
    int w = idx / (KCP - KC), k = KC + (idx - w * (KCP - KC));
    sA[w * KCP + k] = (__bf16)0.0f;             // zero K padding
  }
  __syncthreads();

  // Preload A fragments for both M-tiles (w = l15 and w = 16+l15)
  v16bf afr0[13], afr1[13];
  #pragma unroll
  for (int ks = 0; ks < 13; ++ks) {
    #pragma unroll
    for (int e = 0; e < 16; ++e) {
      int K = ks * 32 + koffA(e, half);
      afr0[ks][e] = sA[l15 * KCP + K];
      afr1[ks][e] = sA[(16 + l15) * KCP + K];
    }
  }

  // ---------- Phase B: per j-tile (13 tiles of 16 output tokens) ----------
  for (int jt = 0; jt < 13; ++jt) {
    const int j0  = jt * 16;
    const int jg  = j0 + l15;                   // this lane's global j (B/C column)
    const int jcl = (jg < HW_) ? jg : (HW_ - 1);

    // ---- scores: per source i, two 16x16 WMMA tiles sharing each B load ----
    for (int i = wave; i < HW_; i += 8) {
      v8f acc0 = {0.f, 0.f, 0.f, 0.f, 0.f, 0.f, 0.f, 0.f};
      v8f acc1 = {0.f, 0.f, 0.f, 0.f, 0.f, 0.f, 0.f, 0.f};
      if (TPOSE) {
        const __bf16* wrow = WgT + (size_t)(i * HW_ + jcl) * KCP + half * 16;
        #pragma unroll
        for (int ks = 0; ks < 13; ++ks) {
          v16bf bfr = *(const v16bf*)(wrow + ks * 32);
          acc0 = __builtin_amdgcn_wmma_f32_16x16x32_bf16(
                     false, afr0[ks], false, bfr, (short)0, acc0, false, false);
          acc1 = __builtin_amdgcn_wmma_f32_16x16x32_bf16(
                     false, afr1[ks], false, bfr, (short)0, acc1, false, false);
        }
      } else {
        const float* wcol = Wg + (size_t)i * HW_ + jcl;   // row stride NG
        #pragma unroll
        for (int ks = 0; ks < 13; ++ks) {
          v16bf bfr;
          #pragma unroll
          for (int e = 0; e < 16; ++e) {
            int K = ks * 32 + koffB(e, half);
            K = (K < KC) ? K : (KC - 1);                  // clamp; A pad is zero anyway
            bfr[e] = to_bf16(wcol[(size_t)K * NG]);
          }
          acc0 = __builtin_amdgcn_wmma_f32_16x16x32_bf16(
                     false, afr0[ks], false, bfr, (short)0, acc0, false, false);
          acc1 = __builtin_amdgcn_wmma_f32_16x16x32_bf16(
                     false, afr1[ks], false, bfr, (short)0, acc1, false, false);
        }
      }
      if (jg < HW_) {
        float bias = bg[(size_t)i * HW_ + jg];
        // [i][j][w] layout: lane's 8 rows per tile are consecutive w -> one b128 store
        v8bf pk0, pk1;
        #pragma unroll
        for (int r = 0; r < 8; ++r) {
          pk0[r] = to_bf16(acc0[r] + bias);               // w = r + 8*half
          pk1[r] = to_bf16(acc1[r] + bias);               // w = 16 + r + 8*half
        }
        __bf16* srow = sSb + i * 512 + l15 * 32 + half * 8;
        *(v8bf*)(srow)      = pk0;
        *(v8bf*)(srow + 16) = pk1;
      }
    }
    __syncthreads();

    // ---- softmax stats over i (512 columns = (j,w); 2 per thread) ----
    #pragma unroll
    for (int c2 = 0; c2 < 2; ++c2) {
      int col = tid + c2 * 256;                           // col = j*32 + w
      if (j0 + (col >> 5) < HW_) {
        float mx = -3.0e38f;
        #pragma unroll 4
        for (int i = 0; i < HW_; ++i) mx = fmaxf(mx, (float)sSb[i * 512 + col]);
        float sm = 0.f;
        #pragma unroll 4
        for (int i = 0; i < HW_; ++i) sm += __expf((float)sSb[i * 512 + col] - mx);
        sMax[col] = mx;
        sRcp[col] = 1.0f / sm;
      } else {
        sMax[col] = 0.f;
        sRcp[col] = 0.f;                                  // kills invalid columns
      }
    }
    __syncthreads();

    // ---- mixing: out[w][c][j] = sum_i xv[w][c][i] * p[w][i][j] ----
    // 64 tasks = 32 windows x 2 channel-tiles; WMMA M=c, K=i (padded 224), N=j
    for (int task = wave; task < 64; task += 8) {
      const int w  = task >> 1, ct = task & 1;
      const int n2 = w >> 3, m = w & 7;
      const int cg = ct * 16 + l15;                       // A row -> channel
      const float mval = sMax[l15 * 32 + w];              // B col -> j stats
      const float rsum = sRcp[l15 * 32 + w];
      v8f acc = {0.f, 0.f, 0.f, 0.f, 0.f, 0.f, 0.f, 0.f};
      #pragma unroll
      for (int ks = 0; ks < 7; ++ks) {
        v16bf av, bv;
        #pragma unroll
        for (int e = 0; e < 16; ++e) {
          int ia = ks * 32 + koffA(e, half);              // A element: (c=cg, i=ia)
          if (ia < HW_) {
            int tok = (ia / WS_) * (N2_ * WS_) + n2 * WS_ + (ia % WS_);
            av[e] = to_bf16(xb[(size_t)tok * DIM + m * C_ + cg]);
          } else av[e] = (__bf16)0.0f;
          int ib = ks * 32 + koffB(e, half);              // B element: (i=ib, j=l15)
          if (ib < HW_) {
            float p = __expf((float)sSb[ib * 512 + l15 * 32 + w] - mval) * rsum;
            bv[e] = to_bf16(p);
          } else bv[e] = (__bf16)0.0f;
        }
        acc = __builtin_amdgcn_wmma_f32_16x16x32_bf16(
                  false, av, false, bv, (short)0, acc, false, false);
      }
      if (jg < HW_) {
        int tokj = (jg / WS_) * (N2_ * WS_) + n2 * WS_ + (jg % WS_);
        float* op = out + ((size_t)b * NTOK + tokj) * DIM + m * C_ + ct * 16;
        #pragma unroll
        for (int r = 0; r < 8; ++r)
          op[r + half * 8] = acc[r];                      // C: M = r (+8 upper half)
      }
    }
    __syncthreads();                                      // sSb reused next j-tile
  }
}

extern "C" void kernel_launch(void* const* d_in, const int* in_sizes, int n_in,
                              void* d_out, int out_size, void* d_ws, size_t ws_size,
                              hipStream_t stream) {
  (void)in_sizes; (void)n_in; (void)out_size;
  const float* x  = (const float*)d_in[0];
  const float* Wc = (const float*)d_in[1];
  const float* bc = (const float*)d_in[2];
  const float* Wg = (const float*)d_in[3];
  const float* bg = (const float*)d_in[4];
  float* out = (float*)d_out;

  if (d_ws != nullptr && ws_size >= WS_NEED) {
    __bf16* WgT = (__bf16*)d_ws;
    wg_transpose_kernel<<<dim3((NG + 31) / 32, (KCP + 31) / 32), dim3(32, 8), 0, stream>>>(Wg, WgT);
    (void)hipFuncSetAttribute((const void*)mixing_attention_kernel<true>,
                              hipFuncAttributeMaxDynamicSharedMemorySize, SMEM_TOTAL);
    mixing_attention_kernel<true><<<dim3(B_), dim3(256), SMEM_TOTAL, stream>>>(
        x, Wc, bc, Wg, WgT, bg, out);
  } else {
    (void)hipFuncSetAttribute((const void*)mixing_attention_kernel<false>,
                              hipFuncAttributeMaxDynamicSharedMemorySize, SMEM_TOTAL);
    mixing_attention_kernel<false><<<dim3(B_), dim3(256), SMEM_TOTAL, stream>>>(
        x, Wc, bc, Wg, (const __bf16*)nullptr, bg, out);
  }
}
